// MLA_80410377716147
// MI455X (gfx1250) — matrix operand order repaired and checked by reading, not compiled, not run on hardware
//
#include <hip/hip_runtime.h>
#include <cmath>

// ---------- problem constants ----------
#define C_B     2
#define C_S     2048
#define C_DIM   2048
#define C_H     16
#define C_QL    1536
#define C_KVL   512
#define C_NOPE  128
#define C_ROPE  64
#define C_QK    192
#define C_VD    128
#define C_BS    (C_B * C_S)

typedef unsigned short bf16_t;
typedef __attribute__((ext_vector_type(16))) __bf16 v16bf;
typedef __attribute__((ext_vector_type(8)))  float  v8f;

__device__ __forceinline__ bf16_t f2bf(float f) {
  unsigned u = __float_as_uint(f);
  unsigned r = (u + 0x7FFFu + ((u >> 16) & 1u)) >> 16;
  return (bf16_t)r;
}

// ---- CDNA5 async global->LDS copy (ASYNCcnt-tracked, no VGPR bounce) ----
// ISA 15.18.3 opcode 98: GLOBAL_LOAD_ASYNC_TO_LDS_B128, GV mode.
// LDS byte address = low 32 bits of the generic shared pointer (ISA 10.2).
__device__ __forceinline__ void async_ld_b128(void* lds, const void* g) {
  unsigned la = (unsigned)(size_t)lds;
  unsigned long long ga = (unsigned long long)(size_t)g;
  asm volatile("global_load_async_to_lds_b128 %0, %1, off" :: "v"(la), "v"(ga) : "memory");
}
// async loads retire in order -> waiting to a nonzero threshold retires the older tile
__device__ __forceinline__ void wait_async0() { asm volatile("s_wait_asynccnt 0x0" ::: "memory"); }
__device__ __forceinline__ void wait_async4() { asm volatile("s_wait_asynccnt 0x4" ::: "memory"); }
__device__ __forceinline__ void wait_async6() { asm volatile("s_wait_asynccnt 0x6" ::: "memory"); }

// ---------- fp32 -> bf16 convert ----------
__global__ void cvt_f32_bf16(const float* __restrict__ in, bf16_t* __restrict__ out, size_t n) {
  size_t i = (size_t)blockIdx.x * blockDim.x + threadIdx.x;
  size_t st = (size_t)gridDim.x * blockDim.x;
  for (; i < n; i += st) out[i] = f2bf(in[i]);
}

// ---------- rmsnorm rows -> bf16 ----------
__global__ void rmsnorm_bf16(const float* __restrict__ in, const float* __restrict__ scale,
                             bf16_t* __restrict__ out, int D) {
  int row = blockIdx.x;
  const float* r = in + (size_t)row * D;
  __shared__ float red[256];
  float s = 0.f;
  for (int i = threadIdx.x; i < D; i += 256) { float v = r[i]; s += v * v; }
  red[threadIdx.x] = s; __syncthreads();
  for (int st = 128; st > 0; st >>= 1) {
    if (threadIdx.x < st) red[threadIdx.x] += red[threadIdx.x + st];
    __syncthreads();
  }
  float inv = rsqrtf(red[0] / (float)D + 1e-6f);
  for (int i = threadIdx.x; i < D; i += 256)
    out[(size_t)row * D + i] = f2bf(r[i] * inv * scale[i]);
}

// ---------- generic bf16 WMMA GEMM, 128x64 C-tile, 4 waves, double-buffered async LDS ----------
// C[m,n] = sum_k A[m,k] * Bm(k,n)
// bT==1 : B row-major [N][K]  (async B128 staging)
// bT==0 : B row-major [K][N]  (scalar transpose staging)
// batching: z1 = z % n1, z2 = z / n1; offsets = z1*s?1 + z2*s?2 (elements)
__launch_bounds__(128)
__global__ void wmma_gemm(const bf16_t* __restrict__ A, long lda, long sA1, long sA2,
                          const bf16_t* __restrict__ B, long ldb, long sB1, long sB2,
                          void* __restrict__ Cv, long ldc, long sC1, long sC2,
                          int K, int n1, int bT, int outBf16) {
  int z = blockIdx.z; int z1 = z % n1; int z2 = z / n1;
  A += (size_t)z1 * sA1 + (size_t)z2 * sA2;
  B += (size_t)z1 * sB1 + (size_t)z2 * sB2;
  size_t cOff = (size_t)z1 * sC1 + (size_t)z2 * sC2;
  int row0 = blockIdx.y * 128;
  int col0 = blockIdx.x * 64;
  __shared__ __align__(16) bf16_t lsA[2][128 * 40];  // stride 40 (80B, 16B-aligned rows)
  __shared__ __align__(16) bf16_t lsB[2][64 * 40];
  int tid = threadIdx.x;
  int lane = tid & 31, w = tid >> 5;
  int hh = lane >> 4, l16 = lane & 15;
  v8f acc[2][4] = {};
  union F { v16bf v; unsigned u[8]; };

  auto stage = [&](int ib, int k0) {
    const bf16_t* gA = A + (size_t)row0 * lda + k0;
    #pragma unroll
    for (int j = 0; j < 4; ++j) {                // 128 rows x 4 B128-chunks
      int ch = tid + j * 128;
      int r = ch >> 2, c = ch & 3;
      async_ld_b128(&lsA[ib][r * 40 + c * 8], gA + (size_t)r * lda + c * 8);
    }
    if (bT) {
      const bf16_t* gB = B + (size_t)col0 * ldb + k0;
      #pragma unroll
      for (int j = 0; j < 2; ++j) {              // 64 rows x 4 B128-chunks
        int ch = tid + j * 128;
        int r = ch >> 2, c = ch & 3;
        async_ld_b128(&lsB[ib][r * 40 + c * 8], gB + (size_t)r * ldb + c * 8);
      }
    } else {
      for (int j = 0; j < 16; ++j) {             // 32x64 transpose staging
        int e = tid + j * 128;
        int kk = e >> 6, n = e & 63;
        lsB[ib][n * 40 + kk] = B[(size_t)(k0 + kk) * ldb + col0 + n];
      }
    }
  };

  stage(0, 0);
  int ib = 0;
  for (int k0 = 0; k0 < K; k0 += 32) {
    bool hasNext = (k0 + 32) < K;
    if (hasNext) stage(ib ^ 1, k0 + 32);         // overlap next-tile DMA with compute
    if (k0 + 64 < K)
      __builtin_prefetch(A + (size_t)(row0 + tid) * lda + k0 + 64, 0, 0);
    if (hasNext) { if (bT) wait_async6(); else wait_async4(); }
    else         wait_async0();
    __syncthreads();                             // tile ib fully resident

    F fa0, fa1;
    {
      const unsigned* r0 = (const unsigned*)&lsA[ib][(w * 32 + l16) * 40];
      const unsigned* r1 = (const unsigned*)&lsA[ib][(w * 32 + 16 + l16) * 40];
      #pragma unroll
      for (int i = 0; i < 8; ++i) {
        int ix = (i & 3) + ((i >> 2) << 3) + (hh << 2);
        fa0.u[i] = r0[ix];
        fa1.u[i] = r1[ix];
      }
    }
    F fb[4];
    #pragma unroll
    for (int nt = 0; nt < 4; ++nt) {
      const unsigned* rB = (const unsigned*)&lsB[ib][(nt * 16 + l16) * 40];
      #pragma unroll
      for (int i = 0; i < 8; ++i) fb[nt].u[i] = rB[(i & 3) + ((i >> 2) << 3) + (hh << 2)];
    }
    #pragma unroll
    for (int nt = 0; nt < 4; ++nt) {
      acc[0][nt] = __builtin_amdgcn_wmma_f32_16x16x32_bf16(
          false, fa0.v, false, fb[nt].v, (short)0, acc[0][nt], false, false);
      acc[1][nt] = __builtin_amdgcn_wmma_f32_16x16x32_bf16(
          false, fa1.v, false, fb[nt].v, (short)0, acc[1][nt], false, false);
    }
    __syncthreads();                             // all readers done before buffer reuse
    ib ^= 1;
  }
  #pragma unroll
  for (int af = 0; af < 2; ++af)
    #pragma unroll
    for (int nt = 0; nt < 4; ++nt)
      #pragma unroll
      for (int r = 0; r < 8; ++r) {
        int row = row0 + w * 32 + af * 16 + r + 8 * hh;
        int col = col0 + nt * 16 + l16;
        if (outBf16) ((bf16_t*)Cv)[cOff + (size_t)row * ldc + col] = f2bf(acc[af][nt][r]);
        else         ((float*)Cv)[cOff + (size_t)row * ldc + col] = acc[af][nt][r];
      }
}

// ---------- kv prep: rmsnorm(512) + rope(64); writes K-cache and transposed V-cache ----------
__global__ void prep_kv(const float* __restrict__ kv, const float* __restrict__ kvns,
                        const float* __restrict__ freqs,
                        bf16_t* __restrict__ kv_cat, bf16_t* __restrict__ v_t) {
  int tok = blockIdx.x;
  int b = tok / C_S, s = tok % C_S;
  const float* r = kv + (size_t)tok * 576;
  __shared__ float red[256];
  float a = 0.f;
  for (int i = threadIdx.x; i < 512; i += 256) { float v = r[i]; a += v * v; }
  red[threadIdx.x] = a; __syncthreads();
  for (int st = 128; st > 0; st >>= 1) {
    if (threadIdx.x < st) red[threadIdx.x] += red[threadIdx.x + st];
    __syncthreads();
  }
  float inv = rsqrtf(red[0] / 512.f + 1e-6f);
  for (int i = threadIdx.x; i < 512; i += 256) {
    bf16_t bv = f2bf(r[i] * inv * kvns[i]);
    kv_cat[(size_t)tok * 576 + i] = bv;
    v_t[((size_t)b * 512 + i) * C_S + s] = bv;
  }
  if (threadIdx.x < 32) {
    int i = threadIdx.x;
    float c  = freqs[((size_t)s * 32 + i) * 2 + 0];
    float sn = freqs[((size_t)s * 32 + i) * 2 + 1];
    float x0 = r[512 + 2 * i], x1 = r[512 + 2 * i + 1];
    kv_cat[(size_t)tok * 576 + 512 + 2 * i]     = f2bf(x0 * c - x1 * sn);
    kv_cat[(size_t)tok * 576 + 512 + 2 * i + 1] = f2bf(x0 * sn + x1 * c);
  }
}

// ---------- q prep ----------
__global__ void prep_q(const float* __restrict__ q, const float* __restrict__ freqs,
                       bf16_t* __restrict__ q_nope, bf16_t* __restrict__ q_cat) {
  int tok = blockIdx.x;
  int b = tok / C_S, s = tok % C_S;
  const float* r = q + (size_t)tok * (C_H * C_QK);
  for (int i = threadIdx.x; i < C_H * C_NOPE; i += 256) {
    int h = i >> 7, d = i & 127;
    q_nope[((size_t)h * C_BS + tok) * C_NOPE + d] = f2bf(r[h * C_QK + d]);
  }
  for (int i = threadIdx.x; i < C_H * 32; i += 256) {
    int h = i >> 5, p = i & 31;
    float c  = freqs[((size_t)s * 32 + p) * 2 + 0];
    float sn = freqs[((size_t)s * 32 + p) * 2 + 1];
    float x0 = r[h * C_QK + C_NOPE + 2 * p], x1 = r[h * C_QK + C_NOPE + 2 * p + 1];
    size_t base = (((size_t)b * C_H + h) * C_S + s) * 576 + 512;
    q_cat[base + 2 * p]     = f2bf(x0 * c - x1 * sn);
    q_cat[base + 2 * p + 1] = f2bf(x0 * sn + x1 * c);
  }
}

// ---------- flash attention over the 576-dim concat latent ----------
__launch_bounds__(256)
__global__ void mla_attn(const bf16_t* __restrict__ q_cat, const bf16_t* __restrict__ kv_cat,
                         const bf16_t* __restrict__ v_t, bf16_t* __restrict__ o_lat, float scale) {
  int q0 = blockIdx.x * 16;
  int h  = blockIdx.y;
  int b  = blockIdx.z;
  int tid = threadIdx.x;
  int w = tid >> 5, lane = tid & 31;
  int hh = lane >> 4, l16 = lane & 15;

  __shared__ __align__(16) bf16_t sQ[16 * 584];   // row = 1168B (16B-aligned)
  __shared__ float  sS[16 * 128];
  __shared__ __align__(16) bf16_t sP[16 * 132];
  __shared__ float  sAlpha[16];
  __shared__ float  sRowL[16];

  const bf16_t* qrow = q_cat + (((size_t)b * C_H + h) * C_S + q0) * 576;
  for (int i = tid; i < 16 * 72; i += 256) {      // 576 bf16 = 72 B128 chunks per row
    int r = i / 72, c = i % 72;
    async_ld_b128(&sQ[r * 584 + c * 8], qrow + (size_t)r * 576 + c * 8);
  }
  wait_async0();
  __syncthreads();

  float rowM = -3.0e38f, rowL = 0.0f;   // live in wave0 lanes 0..15
  v8f acc[4] = {};
  union F { v16bf v; unsigned u[8]; };

  int nChunks = ((q0 + 15) >> 7) + 1;   // 128-wide causal t-chunks
  for (int ch = 0; ch < nChunks; ++ch) {
    int t0 = ch << 7;
    int tcol = t0 + w * 16 + l16;
    // ---- scores: 16x16 tile per wave, K = 576
    v8f sc = {};
    const unsigned* kr = (const unsigned*)(kv_cat + ((size_t)b * C_S + tcol) * 576);
    const unsigned* rA = (const unsigned*)&sQ[l16 * 584];
    for (int kk = 0; kk < 18; ++kk) {
      F fa, fb;
      const unsigned* baseA = rA + kk * 16;
      #pragma unroll
      for (int i = 0; i < 8; ++i)
        fa.u[i] = baseA[(i & 3) + ((i >> 2) << 3) + (hh << 2)];
      const unsigned* baseB = kr + kk * 16;
      #pragma unroll
      for (int i = 0; i < 8; ++i) fb.u[i] = baseB[(i & 3) + ((i >> 2) << 3) + (hh << 2)];
      sc = __builtin_amdgcn_wmma_f32_16x16x32_bf16(
          false, fa.v, false, fb.v, (short)0, sc, false, false);
    }
    #pragma unroll
    for (int r = 0; r < 8; ++r) {
      int m = r + 8 * hh;
      float v = sc[r] * scale;
      if (tcol > q0 + m) v += -1.0e9f;            // causal mask (matches reference)
      sS[m * 128 + w * 16 + l16] = v;
    }
    __syncthreads();
    // ---- online softmax (wave0, one lane per row)
    if (tid < 16) {
      int m = tid;
      float mx = rowM;
      for (int j = 0; j < 128; ++j) mx = fmaxf(mx, sS[m * 128 + j]);
      float alpha = __expf(rowM - mx);
      float lsum = rowL * alpha;
      for (int j = 0; j < 128; ++j) {
        float p = __expf(sS[m * 128 + j] - mx);
        lsum += p;
        sP[m * 132 + j] = f2bf(p);
      }
      rowM = mx; rowL = lsum;
      sAlpha[m] = alpha; sRowL[m] = lsum;
    }
    __syncthreads();
    // ---- rescale O, accumulate P(16x128) x V(128x64-slice)
    float al[8];
    #pragma unroll
    for (int r = 0; r < 8; ++r) al[r] = sAlpha[r + 8 * hh];
    #pragma unroll
    for (int nt = 0; nt < 4; ++nt)
      #pragma unroll
      for (int r = 0; r < 8; ++r) acc[nt][r] *= al[r];
    for (int ks = 0; ks < 4; ++ks) {
      F fa;
      const unsigned* rP = (const unsigned*)&sP[l16 * 132];
      const unsigned* baseP = rP + ks * 16;
      #pragma unroll
      for (int i = 0; i < 8; ++i)
        fa.u[i] = baseP[(i & 3) + ((i >> 2) << 3) + (hh << 2)];
      F fb[4];
      #pragma unroll
      for (int nt = 0; nt < 4; ++nt) {
        int c = w * 64 + nt * 16 + l16;
        const unsigned* vr = (const unsigned*)(v_t + ((size_t)b * 512 + c) * C_S + t0 + ks * 32);
        #pragma unroll
        for (int i = 0; i < 8; ++i) fb[nt].u[i] = vr[(i & 3) + ((i >> 2) << 3) + (hh << 2)];
      }
      #pragma unroll
      for (int nt = 0; nt < 4; ++nt)
        acc[nt] = __builtin_amdgcn_wmma_f32_16x16x32_bf16(
            false, fa.v, false, fb[nt].v, (short)0, acc[nt], false, false);
    }
    __syncthreads();
  }
  // ---- normalize + write o_lat[b][s][h][512] (bf16)
  #pragma unroll
  for (int nt = 0; nt < 4; ++nt) {
    #pragma unroll
    for (int r = 0; r < 8; ++r) {
      int m = r + 8 * hh;
      float o = acc[nt][r] / sRowL[m];
      int c = w * 64 + nt * 16 + l16;
      o_lat[(((size_t)b * C_S + q0 + m) * C_H + h) * 512 + c] = f2bf(o);
    }
  }
}

// ---------- host ----------
extern "C" void kernel_launch(void* const* d_in, const int* in_sizes, int n_in,
                              void* d_out, int out_size, void* d_ws, size_t ws_size,
                              hipStream_t stream) {
  const float* x     = (const float*)d_in[0];
  const float* freqs = (const float*)d_in[2];
  const float* wq_a  = (const float*)d_in[4];
  const float* qns   = (const float*)d_in[5];
  const float* wq_b  = (const float*)d_in[6];
  const float* wkv_a = (const float*)d_in[7];
  const float* kvns  = (const float*)d_in[8];
  const float* wkv_b = (const float*)d_in[9];
  const float* wo    = (const float*)d_in[10];
  float* out = (float*)d_out;

  char* ws = (char*)d_ws;
  size_t off = 0;
  auto alloc = [&](size_t bytes) { size_t o = off; off += (bytes + 255) & ~(size_t)255; return o; };

  bf16_t* xbf    = (bf16_t*)(ws + alloc((size_t)C_BS * C_DIM * 2));
  bf16_t* wqa_b  = (bf16_t*)(ws + alloc((size_t)C_QL * C_DIM * 2));
  bf16_t* wqb_b  = (bf16_t*)(ws + alloc((size_t)C_H * C_QK * C_QL * 2));
  bf16_t* wkva_b = (bf16_t*)(ws + alloc((size_t)(C_KVL + C_ROPE) * C_DIM * 2));
  bf16_t* wkvb_b = (bf16_t*)(ws + alloc((size_t)C_H * 256 * C_KVL * 2));
  bf16_t* wo_b   = (bf16_t*)(ws + alloc((size_t)C_DIM * C_H * C_VD * 2));
  float*  qlat   = (float*)(ws + alloc((size_t)C_BS * C_QL * 4));
  bf16_t* qlatb  = (bf16_t*)(ws + alloc((size_t)C_BS * C_QL * 2));
  float*  qfull  = (float*)(ws + alloc((size_t)C_BS * C_H * C_QK * 4));
  float*  kvfull = (float*)(ws + alloc((size_t)C_BS * 576 * 4));
  bf16_t* kvcat  = (bf16_t*)(ws + alloc((size_t)C_BS * 576 * 2));
  bf16_t* vt     = (bf16_t*)(ws + alloc((size_t)C_B * 512 * C_S * 2));
  bf16_t* qnope  = (bf16_t*)(ws + alloc((size_t)C_H * C_BS * C_NOPE * 2));
  bf16_t* qcat   = (bf16_t*)(ws + alloc((size_t)C_B * C_H * C_S * 576 * 2));
  bf16_t* olat   = (bf16_t*)(ws + alloc((size_t)C_BS * C_H * 512 * 2));
  float*  o2     = (float*)(ws + alloc((size_t)C_BS * C_H * C_VD * 4));
  bf16_t* o2b    = (bf16_t*)(ws + alloc((size_t)C_BS * C_H * C_VD * 2));
  (void)ws_size; (void)n_in; (void)in_sizes; (void)out_size;

  const float scl = (float)((1.0 / sqrt(192.0)) * pow(0.1 * log(2.0) + 1.0, 2.0));

  // converts
  cvt_f32_bf16<<<2048, 256, 0, stream>>>(x,     xbf,    (size_t)C_BS * C_DIM);
  cvt_f32_bf16<<<2048, 256, 0, stream>>>(wq_a,  wqa_b,  (size_t)C_QL * C_DIM);
  cvt_f32_bf16<<<2048, 256, 0, stream>>>(wq_b,  wqb_b,  (size_t)C_H * C_QK * C_QL);
  cvt_f32_bf16<<<2048, 256, 0, stream>>>(wkv_a, wkva_b, (size_t)(C_KVL + C_ROPE) * C_DIM);
  cvt_f32_bf16<<<2048, 256, 0, stream>>>(wkv_b, wkvb_b, (size_t)C_H * 256 * C_KVL);
  cvt_f32_bf16<<<2048, 256, 0, stream>>>(wo,    wo_b,   (size_t)C_DIM * C_H * C_VD);

  // q_lat = x . wq_a^T   [4096 x 1536], K=2048
  wmma_gemm<<<dim3(C_QL / 64, C_BS / 128, 1), 128, 0, stream>>>(
      xbf, C_DIM, 0, 0, wqa_b, C_DIM, 0, 0, qlat, C_QL, 0, 0, C_DIM, 1, 1, 0);
  rmsnorm_bf16<<<C_BS, 256, 0, stream>>>(qlat, qns, qlatb, C_QL);
  // q = q_lat_n . wq_b^T  [4096 x 3072], K=1536
  wmma_gemm<<<dim3(C_H * C_QK / 64, C_BS / 128, 1), 128, 0, stream>>>(
      qlatb, C_QL, 0, 0, wqb_b, C_QL, 0, 0, qfull, C_H * C_QK, 0, 0, C_QL, 1, 1, 0);
  // kv = x . wkv_a^T  [4096 x 576], K=2048
  wmma_gemm<<<dim3(576 / 64, C_BS / 128, 1), 128, 0, stream>>>(
      xbf, C_DIM, 0, 0, wkva_b, C_DIM, 0, 0, kvfull, 576, 0, 0, C_DIM, 1, 1, 0);

  prep_kv<<<C_BS, 256, 0, stream>>>(kvfull, kvns, freqs, kvcat, vt);
  prep_q<<<C_BS, 256, 0, stream>>>(qfull, freqs, qnope, qcat);

  // q_abs: per (h,b): [2048 x 512] = q_nope[h,b] (K=128) . wkv_b_nope[h] (128x512, K-major)
  wmma_gemm<<<dim3(512 / 64, C_S / 128, C_H * C_B), 128, 0, stream>>>(
      qnope, C_NOPE, (long)C_BS * C_NOPE, (long)C_S * C_NOPE,
      wkvb_b, C_KVL, (long)256 * C_KVL, 0,
      qcat, 576, (long)C_S * 576, (long)C_H * C_S * 576,
      C_NOPE, C_H, 0, 1);

  // flash attention
  mla_attn<<<dim3(C_S / 16, C_H, C_B), 256, 0, stream>>>(qcat, kvcat, vt, olat, scl);

  // o2: per (h,b): [2048 x 128] = o_lat[b,:,h,:] (K=512) . wkv_b_v[h]^T (N-major)
  wmma_gemm<<<dim3(C_VD / 64, C_S / 128, C_H * C_B), 128, 0, stream>>>(
      olat, (long)C_H * 512, 512, (long)C_S * C_H * 512,
      wkvb_b + (size_t)C_NOPE * C_KVL, C_KVL, (long)256 * C_KVL, 0,
      o2, (long)C_H * C_VD, C_VD, (long)C_S * C_H * C_VD,
      512, C_H, 1, 0);

  cvt_f32_bf16<<<2048, 256, 0, stream>>>(o2, o2b, (size_t)C_BS * C_H * C_VD);

  // out = o2 . wo^T  [4096 x 2048], K=2048
  wmma_gemm<<<dim3(C_DIM / 64, C_BS / 128, 1), 128, 0, stream>>>(
      o2b, C_H * C_VD, 0, 0, wo_b, C_H * C_VD, 0, 0, out, C_DIM, 0, 0, C_H * C_VD, 1, 1, 0);
}
